// AttentionAspect_48541720379808
// MI455X (gfx1250) — matrix operand order repaired and checked by reading, not compiled, not run on hardware
//
#include <hip/hip_runtime.h>
#include <math.h>

typedef __attribute__((ext_vector_type(16))) __bf16 v16bf;
typedef __attribute__((ext_vector_type(8)))  __bf16 v8bf;
typedef __attribute__((ext_vector_type(4)))  __bf16 v4bf;
typedef __attribute__((ext_vector_type(8)))  float  v8f;
typedef __attribute__((ext_vector_type(4)))  int    v4i;

#define BM 128
#define BN 128
#define BK 32
#define LDT 40   // LDS row stride in bf16 elements (80B: 16B-aligned rows, conflict padding)

#if defined(__has_builtin)
#if __has_builtin(__builtin_amdgcn_global_load_async_to_lds_b128) && \
    __has_builtin(__builtin_amdgcn_s_wait_asynccnt)
#define HAVE_ASYNC 1
#endif
#endif
#ifndef HAVE_ASYNC
#define HAVE_ASYNC 0
#endif

static __device__ __forceinline__ v16bf cat8(v8bf lo, v8bf hi) {
  return __builtin_shufflevector(lo, hi, 0,1,2,3,4,5,6,7,8,9,10,11,12,13,14,15);
}

#if HAVE_ASYNC
// 16-byte async global -> LDS copy (gfx1250 GLOBAL_LOAD_ASYNC_TO_LDS_B128, ASYNCcnt)
static __device__ __forceinline__ void async_cp16(const void* g, void* l) {
  __builtin_amdgcn_global_load_async_to_lds_b128(
      (__attribute__((address_space(1))) v4i*)g,
      (__attribute__((address_space(3))) v4i*)l,
      0, 0);
}
#endif

// ---------------------------------------------------------------------------
// Unified bf16-WMMA GEMM.
//   C[M,N] = A[M,K] * B^T   where B is stored as Bs[N][K] (row-major, ldb=K stride)
// A_F32: A is f32 in global (converted to bf16 while staging into LDS), else bf16.
// EPI:   0 = store bf16 (workspace), 1 = store f32 * scale, 2 = store f32 + bias[col]
// Batch: two batch dims (blockIdx.y, blockIdx.z) with independent strides per operand.
// ---------------------------------------------------------------------------
template<int A_F32, int EPI>
__global__ __launch_bounds__(256)
void gemm_wmma(const void* __restrict__ Ab_, const __bf16* __restrict__ Bb_,
               void* __restrict__ Cb_,
               int K, int lda, int ldb, int ldc, int tilesN,
               long sAy, long sAz, long sBy, long sBz, long sCy, long sCz,
               float scale, const float* __restrict__ bias)
{
  __shared__ __bf16 As[BM * LDT];
  __shared__ __bf16 Bs[BN * LDT];

  const int zy = blockIdx.y, zz = blockIdx.z;
  const int tm = blockIdx.x / tilesN;
  const int tn = blockIdx.x % tilesN;
  const int tid  = threadIdx.x;
  const int lane = tid & 31;
  const int wave = tid >> 5;
  const int wm = wave & 3;   // 4 waves along M  (32 rows each)
  const int wn = wave >> 2;  // 2 waves along N  (64 cols each)

  const long aoff = zy * sAy + zz * sAz;
  const float*  Af = (const float*)Ab_  + aoff;
  const __bf16* Ah = (const __bf16*)Ab_ + aoff;
  const __bf16* B  = Bb_ + zy * sBy + zz * sBz;
  const long coff  = zy * sCy + zz * sCz;

  const int row0 = tm * BM;
  const int col0 = tn * BN;

  v8f zero = {0.f,0.f,0.f,0.f,0.f,0.f,0.f,0.f};
  v8f acc[2][4];
  for (int i = 0; i < 2; ++i)
    for (int j = 0; j < 4; ++j) acc[i][j] = zero;

  const int half = (lane >= 16) ? 1 : 0;
  const int l16  = lane & 15;

  for (int kk = 0; kk < K; kk += BK) {
    // ---- stage A tile (BM x BK) into LDS as bf16 ----
    if (A_F32) {
      const float* Ap = Af + (long)row0 * lda + kk;
      // prefetch next K-tile of A while we convert this one
      if (kk + BK < K) {
        const float* Pn = Af + (long)(row0 + (tid >> 1)) * lda + kk + BK + (tid & 1) * 16;
        __builtin_prefetch(Pn, 0, 1);
      }
      #pragma unroll
      for (int i = 0; i < 4; ++i) {
        int f  = tid + i * 256;        // 1024 float4 chunks
        int r  = f >> 3;
        int c4 = (f & 7) * 4;
        float4 v = *(const float4*)(Ap + (long)r * lda + c4);
        v4bf o; o.x = (__bf16)v.x; o.y = (__bf16)v.y; o.z = (__bf16)v.z; o.w = (__bf16)v.w;
        *(v4bf*)&As[r * LDT + c4] = o;
      }
    } else {
      const __bf16* Ap = Ah + (long)row0 * lda + kk;
      #pragma unroll
      for (int i = 0; i < 2; ++i) {
        int f  = tid + i * 256;        // 512 chunks of 8 bf16
        int r  = f >> 2;
        int c8 = (f & 3) * 8;
#if HAVE_ASYNC
        async_cp16(Ap + (long)r * lda + c8, &As[r * LDT + c8]);
#else
        *(uint4*)&As[r * LDT + c8] = *(const uint4*)(Ap + (long)r * lda + c8);
#endif
      }
    }
    // ---- stage B tile (BN x BK) into LDS (already bf16, rows contiguous in K) ----
    {
      const __bf16* Bp = B + (long)col0 * ldb + kk;
      #pragma unroll
      for (int i = 0; i < 2; ++i) {
        int f  = tid + i * 256;
        int r  = f >> 2;
        int c8 = (f & 3) * 8;
#if HAVE_ASYNC
        async_cp16(Bp + (long)r * ldb + c8, &Bs[r * LDT + c8]);
#else
        *(uint4*)&Bs[r * LDT + c8] = *(const uint4*)(Bp + (long)r * ldb + c8);
#endif
      }
    }
#if HAVE_ASYNC
    __builtin_amdgcn_s_wait_asynccnt(0);
#endif
    __syncthreads();

    // ---- build fragments per ISA VGPR layouts ----
    // A 16x32: lane<16 -> M=lane,   e0..7 = K0..7,  e8..15 = K16..23
    //          lane>=16-> M=lane-16,e0..7 = K8..15, e8..15 = K24..31
    v16bf afr[2];
    #pragma unroll
    for (int i = 0; i < 2; ++i) {
      int r = wm * 32 + i * 16 + l16;
      v8bf lo = *(const v8bf*)&As[r * LDT + half * 8];
      v8bf hi = *(const v8bf*)&As[r * LDT + 16 + half * 8];
      afr[i] = cat8(lo, hi);
    }
    // B 32x16: lane%16 = N column, e0..7 = K(lane/16)*8.., e8..15 = K16+(lane/16)*8.. (same as A)
    v16bf bfr[4];
    #pragma unroll
    for (int j = 0; j < 4; ++j) {
      int n = wn * 64 + j * 16 + l16;
      v8bf lo = *(const v8bf*)&Bs[n * LDT + half * 8];
      v8bf hi = *(const v8bf*)&Bs[n * LDT + 16 + half * 8];
      bfr[j] = cat8(lo, hi);
    }

    #pragma unroll
    for (int i = 0; i < 2; ++i)
      #pragma unroll
      for (int j = 0; j < 4; ++j)
        acc[i][j] = __builtin_amdgcn_wmma_f32_16x16x32_bf16(
            false, afr[i], false, bfr[j], (short)0, acc[i][j], false, false);

    __syncthreads();
  }

  // ---- epilogue: v8f element g maps to (M = g + half*8, N = lane%16) ----
  #pragma unroll
  for (int i = 0; i < 2; ++i) {
    #pragma unroll
    for (int j = 0; j < 4; ++j) {
      int rbase = row0 + wm * 32 + i * 16 + half * 8;
      int cbase = col0 + wn * 64 + j * 16 + l16;
      #pragma unroll
      for (int g = 0; g < 8; ++g) {
        float v = acc[i][j][g];
        long idx = coff + (long)(rbase + g) * ldc + cbase;
        if (EPI == 0)      ((__bf16*)Cb_)[idx] = (__bf16)v;
        else if (EPI == 1) ((float*)Cb_)[idx]  = v * scale;
        else               ((float*)Cb_)[idx]  = v + bias[cbase];
      }
    }
  }
}

// ---------------------------------------------------------------------------
// Prep kernels
// ---------------------------------------------------------------------------

// w (H,1024,128) f32  ->  wt (H,128,1024) bf16   (transpose + convert)
__global__ void wtrans_cvt_kernel(const float* __restrict__ w, __bf16* __restrict__ wt) {
  int o = blockIdx.x * 256 + threadIdx.x;        // H*128*1024 = 1048576
  int h  = o >> 17;
  int n  = (o >> 10) & 127;
  int kk = o & 1023;
  wt[o] = (__bf16)w[h * 131072 + kk * 128 + n];
}

// plain f32 -> bf16 convert
__global__ void cvt_bf16_kernel(const float* __restrict__ in, __bf16* __restrict__ out, int n) {
  int o = blockIdx.x * 256 + threadIdx.x;
  if (o < n) out[o] = (__bf16)in[o];
}

// kx (256,512,128) bf16 -> kxT (256,128,512) bf16
__global__ void kx_trans_kernel(const __bf16* __restrict__ kx, __bf16* __restrict__ kxT) {
  int o = blockIdx.x * 256 + threadIdx.x;        // 256*128*512 = 16777216
  int z = o >> 16;
  int n = (o >> 9) & 127;
  int kp = o & 511;
  kxT[o] = kx[z * 65536 + kp * 128 + n];
}

// in-place softmax over rows of 512 f32
__global__ __launch_bounds__(256) void softmax512_kernel(float* __restrict__ s) {
  float* row = s + (long)blockIdx.x * 512;
  __shared__ float red[256];
  int t = threadIdx.x;
  float a = row[t], b = row[t + 256];
  red[t] = fmaxf(a, b);
  __syncthreads();
  for (int o = 128; o > 0; o >>= 1) { if (t < o) red[t] = fmaxf(red[t], red[t + o]); __syncthreads(); }
  float m = red[0];
  __syncthreads();
  float ea = __expf(a - m), eb = __expf(b - m);
  red[t] = ea + eb;
  __syncthreads();
  for (int o = 128; o > 0; o >>= 1) { if (t < o) red[t] += red[t + o]; __syncthreads(); }
  float inv = 1.0f / red[0];
  row[t] = ea * inv;
  row[t + 256] = eb * inv;
}

// ---------------------------------------------------------------------------
extern "C" void kernel_launch(void* const* d_in, const int* in_sizes, int n_in,
                              void* d_out, int out_size, void* d_ws, size_t ws_size,
                              hipStream_t stream) {
  const float* k      = (const float*)d_in[0];   // (32,512,1024)
  const float* q      = (const float*)d_in[1];   // (32,512,1024)
  const float* w_kx   = (const float*)d_in[2];   // (8,1024,128)
  const float* w_qx   = (const float*)d_in[3];   // (8,1024,128)
  const float* proj_w = (const float*)d_in[4];   // (1024,1024)
  const float* proj_b = (const float*)d_in[5];   // (1024)

  float* out   = (float*)d_out;                  // (32,512,1024)
  float* score = out + (long)32 * 512 * 1024;    // (256,512,512)

  char* ws = (char*)d_ws;
  const size_t SZ = 33554432;                    // 16M bf16 elements
  __bf16* kx  = (__bf16*)(ws);                   // (256,512,128)
  __bf16* qx  = (__bf16*)(ws + SZ);              // (256,512,128)
  __bf16* kxT = (__bf16*)(ws + 2 * SZ);          // (256,128,512)
  __bf16* ctx = (__bf16*)(ws + 3 * SZ);          // (32,512,8,128) = (16384,1024)
  __bf16* wkT = (__bf16*)(ws + 4 * SZ);          // (8,128,1024)
  __bf16* wqT = (__bf16*)(ws + 4 * SZ + 2097152);
  __bf16* pw  = (__bf16*)(ws + 4 * SZ + 2 * 2097152);  // (1024,1024) = [N][K]

  const float inv_sqrt_d = 0.08838834764831845f; // 1/sqrt(128)

  // ---- prep: weight transpose/convert ----
  wtrans_cvt_kernel<<<1048576 / 256, 256, 0, stream>>>(w_kx, wkT);
  wtrans_cvt_kernel<<<1048576 / 256, 256, 0, stream>>>(w_qx, wqT);
  cvt_bf16_kernel<<<1048576 / 256, 256, 0, stream>>>(proj_w, pw, 1048576);

  // ---- kx = K @ Wk : A=k[b] f32 (M=512,K=1024), B=wkT[h] (N=128), C=kx[h*32+b] bf16 ----
  gemm_wmma<1, 0><<<dim3(4, 32, 8), 256, 0, stream>>>(
      k, wkT, kx, 1024, 1024, 1024, 128, 1,
      524288L, 0L, 0L, 131072L, 65536L, 2097152L, 1.f, nullptr);
  // ---- qx = Q @ Wq ----
  gemm_wmma<1, 0><<<dim3(4, 32, 8), 256, 0, stream>>>(
      q, wqT, qx, 1024, 1024, 1024, 128, 1,
      524288L, 0L, 0L, 131072L, 65536L, 2097152L, 1.f, nullptr);

  // ---- score = (qx . kx^T) / sqrt(128) : A=qx[z] (M=512,K=128), Bs=kx[z] (N=512), f32 out ----
  gemm_wmma<0, 1><<<dim3(16, 256, 1), 256, 0, stream>>>(
      qx, kx, score, 128, 128, 128, 512, 4,
      65536L, 0L, 65536L, 0L, 262144L, 0L, inv_sqrt_d, nullptr);

  // ---- softmax rows (256*512 rows of 512) ----
  softmax512_kernel<<<131072, 256, 0, stream>>>(score);

  // ---- kxT for ctx GEMM ----
  kx_trans_kernel<<<65536, 256, 0, stream>>>(kx, kxT);

  // ---- ctx = score . kx : A=score[z] f32 (M=512,K=512), Bs=kxT[z] (N=128),
  //      C written bf16 at ctx[b][m][h][:] (ldc=1024, base offset b*524288 + h*128) ----
  gemm_wmma<1, 0><<<dim3(4, 32, 8), 256, 0, stream>>>(
      score, kxT, ctx, 512, 512, 512, 1024, 1,
      262144L, 8388608L, 65536L, 2097152L, 524288L, 128L, 1.f, nullptr);

  // ---- out = ctx @ proj_w^T + b : A=ctx bf16 (M=16384,K=1024), Bs=pw (N=1024), f32 + bias ----
  gemm_wmma<0, 2><<<dim3(1024, 1, 1), 256, 0, stream>>>(
      ctx, pw, out, 1024, 1024, 1024, 1024, 8,
      0L, 0L, 0L, 0L, 0L, 0L, 1.f, proj_b);
}
